// NeuralODE_63075889709482
// MI455X (gfx1250) — matrix operand order repaired and checked by reading, not compiled, not run on hardware
//
#include <hip/hip_runtime.h>
#include <math.h>

#define DATA_N 2048
#define WIDTH_N 4096
#define NWG 256
#define TPB 256
#define NWAVES 8
#define PFD 4   // A-fragment prefetch queue depth (loads in flight per wave)

typedef __attribute__((ext_vector_type(16))) __bf16   v16bf;
typedef __attribute__((ext_vector_type(8)))  float    v8f;
typedef __attribute__((ext_vector_type(4)))  unsigned u32x4;

// Two b128 loads = one 32B A fragment, issued via asm so the compiler cannot
// collapse the prefetch queue into a single register quad + wait(0).
#define A_LOAD(dlo, dhi, ptr)                                   \
  asm volatile("global_load_b128 %0, %2, off\n\t"               \
               "global_load_b128 %1, %2, off offset:16"         \
               : "=v"(dlo), "=v"(dhi) : "v"(ptr) : "memory")

// Wait until fragment (dlo,dhi) is resident: loads complete in order, and at
// most PFD-1 = 3 newer fragments (6 b128 loads) are in flight behind it.
// Tied operands make the consuming WMMA data-dependent on this wait.
#define A_WAIT(dlo, dhi)                                        \
  asm volatile("s_wait_loadcnt 0x6" : "+v"(dlo), "+v"(dhi))

__device__ __forceinline__ unsigned short f32_to_bf16_bits(float x) {
  unsigned u = __float_as_uint(x);
  unsigned r = u + 0x7fffu + ((u >> 16) & 1u);   // round-to-nearest-even
  return (unsigned short)(r >> 16);
}

__device__ __forceinline__ float softplus_f(float x) {
  return fmaxf(x, 0.0f) + log1pf(expf(-fabsf(x)));   // stable log(1+e^x)
}

// Sense-reversing grid-wide barrier (persistent kernel; NWG workgroups resident).
__device__ __forceinline__ void grid_barrier(unsigned* cnt, unsigned* gen, unsigned nwg) {
  __threadfence();
  __syncthreads();
  if (threadIdx.x == 0) {
    unsigned g = __hip_atomic_load(gen, __ATOMIC_RELAXED, __HIP_MEMORY_SCOPE_AGENT);
    unsigned arrived = __hip_atomic_fetch_add(cnt, 1u, __ATOMIC_ACQ_REL, __HIP_MEMORY_SCOPE_AGENT);
    if (arrived == nwg - 1u) {
      __hip_atomic_store(cnt, 0u, __ATOMIC_RELAXED, __HIP_MEMORY_SCOPE_AGENT);
      __hip_atomic_store(gen, g + 1u, __ATOMIC_RELEASE, __HIP_MEMORY_SCOPE_AGENT);
    } else {
      while (__hip_atomic_load(gen, __ATOMIC_ACQUIRE, __HIP_MEMORY_SCOPE_AGENT) == g) {
        __builtin_amdgcn_s_sleep(1);
      }
    }
  }
  __syncthreads();
  __threadfence();
}

// out[m] = act( sum_k W[m,k]*x[k] + bias[m] ), W pre-packed in WMMA A-fragment
// order (tile = 16 rows x 32 K = 32 lanes x 32B). x is staged into LDS so B
// reads use DScnt and never drain the global-load (A) pipeline.
__device__ __forceinline__ void gemv_layer(
    const unsigned short* __restrict__ Wp,
    const float* __restrict__ bias,
    const unsigned short* __restrict__ xbf,
    unsigned short* __restrict__ outbf,
    int M, int K, float* red, unsigned short* lds_x,
    float* yf, unsigned short* ybf, float* ysOut, float dt)
{
  const int lane = threadIdx.x & 31;
  const int wave = threadIdx.x >> 5;
  const int hi   = lane >> 4;
  const int ktiles = K >> 5;           // K / 32
  const int ksteps = ktiles / NWAVES;  // WMMA steps per wave (8 or 16; >= PFD)
  const int ntiles = M >> 4;           // M / 16

  // Stage input vector into LDS (K*2 bytes, 16B vectorized).
  for (int i = threadIdx.x; i < (K >> 3); i += TPB)
    ((u32x4*)lds_x)[i] = ((const u32x4*)xbf)[i];
  __syncthreads();

  for (int tile = blockIdx.x; tile < ntiles; tile += NWG) {
    v8f c = {0.f, 0.f, 0.f, 0.f, 0.f, 0.f, 0.f, 0.f};
    const int kt0 = wave * ksteps;
    // per-lane fragment base: 512 ushort per tile, 32B per lane
    const unsigned short* aptr =
        Wp + ((size_t)tile * ktiles + kt0) * 512 + (size_t)lane * 16;

    u32x4 qlo[PFD], qhi[PFD];
    #pragma unroll
    for (int p = 0; p < PFD; ++p)
      A_LOAD(qlo[p], qhi[p], aptr + (size_t)p * 512);

    #pragma unroll
    for (int s = 0; s < ksteps; ++s) {
      const int slot = s & (PFD - 1);
      A_WAIT(qlo[slot], qhi[slot]);
      union { u32x4 q[2]; v16bf v; } u;
      u.q[0] = qlo[slot];
      u.q[1] = qhi[slot];
      // B fragment from LDS: lanes 0-15 hold K=k0..k0+15, lanes 16-31 K=+16..+31,
      // all 16 columns replicated with the x chunk.
      const v16bf* bl = (const v16bf*)(lds_x + ((kt0 + s) << 5));
      v16bf b = bl[hi];
      c = __builtin_amdgcn_wmma_f32_16x16x32_bf16(false, u.v, false, b,
                                                  (short)0, c, false, false);
      if (s + PFD < ksteps)
        A_LOAD(qlo[slot], qhi[slot], aptr + (size_t)(s + PFD) * 512);
    }

    // Lane L holds rows m = tile*16 + r + 8*hi (replicated over 16 columns).
    if (lane == 0) {
      #pragma unroll
      for (int i = 0; i < 8; ++i) red[wave * 16 + i] = c[i];
    } else if (lane == 16) {
      #pragma unroll
      for (int i = 0; i < 8; ++i) red[wave * 16 + 8 + i] = c[i];
    }
    __syncthreads();
    if (threadIdx.x < 16) {
      float s = 0.f;
      #pragma unroll
      for (int w = 0; w < NWAVES; ++w) s += red[w * 16 + threadIdx.x];
      int m = tile * 16 + threadIdx.x;
      s += bias[m];
      if (ysOut == nullptr) {
        outbf[m] = f32_to_bf16_bits(softplus_f(s));   // hidden layer
      } else {
        float yn = yf[m] + dt * s;                    // Euler step
        yf[m] = yn;
        ybf[m] = f32_to_bf16_bits(yn);
        ysOut[m] = yn;
      }
    }
    __syncthreads();
  }
}

__global__ __launch_bounds__(TPB) void ode_persistent(
    const float* __restrict__ ts,
    const float* __restrict__ b0, const float* __restrict__ b1,
    const float* __restrict__ b2, const float* __restrict__ b3,
    const unsigned short* __restrict__ W0p, const unsigned short* __restrict__ W1p,
    const unsigned short* __restrict__ W2p, const unsigned short* __restrict__ W3p,
    float* __restrict__ ys, float* __restrict__ yf, unsigned short* __restrict__ ybf,
    unsigned short* __restrict__ h1, unsigned short* __restrict__ h2,
    unsigned short* __restrict__ h3,
    unsigned* barCnt, unsigned* barGen, int T)
{
  __shared__ float red[NWAVES * 16];
  __shared__ unsigned short lds_x[WIDTH_N];
  const float dt = ts[1] - ts[0];
  for (int t = 0; t < T; ++t) {
    gemv_layer(W0p, b0, ybf, h1, WIDTH_N, DATA_N,  red, lds_x, nullptr, nullptr, nullptr, 0.f);
    grid_barrier(barCnt, barGen, NWG);
    gemv_layer(W1p, b1, h1,  h2, WIDTH_N, WIDTH_N, red, lds_x, nullptr, nullptr, nullptr, 0.f);
    grid_barrier(barCnt, barGen, NWG);
    gemv_layer(W2p, b2, h2,  h3, WIDTH_N, WIDTH_N, red, lds_x, nullptr, nullptr, nullptr, 0.f);
    grid_barrier(barCnt, barGen, NWG);
    gemv_layer(W3p, b3, h3, nullptr, DATA_N, WIDTH_N, red, lds_x,
               yf, ybf, ys + (size_t)t * DATA_N, dt);
    grid_barrier(barCnt, barGen, NWG);
  }
}

// fp32 row-major W[M,K] -> bf16 packed in WMMA 16-bit A-fragment order.
__global__ void pack_weights(const float* __restrict__ W, unsigned short* __restrict__ out,
                             int M, int K)
{
  const int ktiles = K >> 5;
  const size_t total = (size_t)M * (size_t)K;
  for (size_t e = (size_t)blockIdx.x * blockDim.x + threadIdx.x; e < total;
       e += (size_t)gridDim.x * blockDim.x) {
    size_t tile = e >> 9;
    int within = (int)(e & 511);
    int lane = within >> 4, i = within & 15;
    int mt = (int)(tile / ktiles), kt = (int)(tile % ktiles);
    int m = mt * 16 + (lane & 15);
    int k = kt * 32 + (lane >> 4) * 16 + i;
    out[e] = f32_to_bf16_bits(W[(size_t)m * K + k]);
  }
}

__global__ void init_state(const float* __restrict__ y0, float* __restrict__ yf,
                           unsigned short* __restrict__ ybf,
                           unsigned* barCnt, unsigned* barGen)
{
  int i = blockIdx.x * blockDim.x + threadIdx.x;
  if (i < DATA_N) {
    float v = y0[i];
    yf[i] = v;
    ybf[i] = f32_to_bf16_bits(v);
  }
  if (i == 0) { *barCnt = 0u; *barGen = 0u; }
}

extern "C" void kernel_launch(void* const* d_in, const int* in_sizes, int n_in,
                              void* d_out, int out_size, void* d_ws, size_t ws_size,
                              hipStream_t stream) {
  const float* ts = (const float*)d_in[0];
  const float* y0 = (const float*)d_in[1];
  const float* W0 = (const float*)d_in[2];
  const float* b0 = (const float*)d_in[3];
  const float* W1 = (const float*)d_in[4];
  const float* b1 = (const float*)d_in[5];
  const float* W2 = (const float*)d_in[6];
  const float* b2 = (const float*)d_in[7];
  const float* W3 = (const float*)d_in[8];
  const float* b3 = (const float*)d_in[9];
  const int T = in_sizes[0];

  char* ws = (char*)d_ws;
  size_t off = 0;
  unsigned short* W0p = (unsigned short*)(ws + off); off += (size_t)WIDTH_N * DATA_N  * 2;
  unsigned short* W1p = (unsigned short*)(ws + off); off += (size_t)WIDTH_N * WIDTH_N * 2;
  unsigned short* W2p = (unsigned short*)(ws + off); off += (size_t)WIDTH_N * WIDTH_N * 2;
  unsigned short* W3p = (unsigned short*)(ws + off); off += (size_t)DATA_N  * WIDTH_N * 2;
  float*          yf  = (float*)(ws + off);          off += (size_t)DATA_N * 4;
  unsigned short* ybf = (unsigned short*)(ws + off); off += (size_t)DATA_N * 2;
  unsigned short* h1  = (unsigned short*)(ws + off); off += (size_t)WIDTH_N * 2;
  unsigned short* h2  = (unsigned short*)(ws + off); off += (size_t)WIDTH_N * 2;
  unsigned short* h3  = (unsigned short*)(ws + off); off += (size_t)WIDTH_N * 2;
  unsigned*   barCnt  = (unsigned*)(ws + off);       off += 128;
  unsigned*   barGen  = (unsigned*)(ws + off);       off += 128;
  if (ws_size < off) return;  // workspace too small; avoid corruption

  // One-time (per call) weight conversion fp32 -> packed bf16 (~100 MB, fits L2).
  pack_weights<<<4096, 256, 0, stream>>>(W0, W0p, WIDTH_N, DATA_N);
  pack_weights<<<4096, 256, 0, stream>>>(W1, W1p, WIDTH_N, WIDTH_N);
  pack_weights<<<4096, 256, 0, stream>>>(W2, W2p, WIDTH_N, WIDTH_N);
  pack_weights<<<4096, 256, 0, stream>>>(W3, W3p, DATA_N,  WIDTH_N);
  init_state<<<(DATA_N + 255) / 256, 256, 0, stream>>>(y0, yf, ybf, barCnt, barGen);

  ode_persistent<<<NWG, TPB, 0, stream>>>(ts, b0, b1, b2, b3,
                                          W0p, W1p, W2p, W3p,
                                          (float*)d_out, yf, ybf, h1, h2, h3,
                                          barCnt, barGen, T);
}
